// IdBce_24696061952158
// MI455X (gfx1250) — matrix operand order repaired and checked by reading, not compiled, not run on hardware
//
#include <hip/hip_runtime.h>
#include <hip/hip_bf16.h>

// ---------------------------------------------------------------------------
// Problem constants (match reference)
// ---------------------------------------------------------------------------
#define B_ROWS 8192
#define D_DIM  512
// GEMM tiling
#define BM   128           // block tile (rows == cols)
#define KC   64            // K chunk staged in LDS
#define LDA  72            // KC + 8 bf16 pad -> 144B row stride (16B aligned, bank-spread)
#define NCHUNK (D_DIM / KC)               // 8
#define SEG_PER_ROW (KC / 8)              // 8 x 16B segments per row
#define SEGS (BM * SEG_PER_ROW)           // 1024 segments per tile per chunk
#define SEGS_PER_THREAD (SEGS / 256)      // 4

typedef __bf16 v16bf __attribute__((ext_vector_type(16)));
typedef float  v8f   __attribute__((ext_vector_type(8)));
typedef unsigned int u32x4 __attribute__((ext_vector_type(4)));
typedef int          i32x4 __attribute__((ext_vector_type(4)));
typedef int          i32x8 __attribute__((ext_vector_type(8)));

struct Bf16x16Bits { uint4 lo, hi; };   // 32 bytes == v16bf

// Build a WMMA 16-bit fragment from two 16B LDS chunks (two ds_load_b128).
__device__ __forceinline__ v16bf load_frag(const unsigned short* p_lo,
                                           const unsigned short* p_hi) {
  Bf16x16Bits t;
  t.lo = *(const uint4*)p_lo;   // fragment elements 0..7  (VGPR 0-3)
  t.hi = *(const uint4*)p_hi;   // fragment elements 8..15 (VGPR 4-7)
  return __builtin_bit_cast(v16bf, t);
}

// ---------------------------------------------------------------------------
// Tensor Data Mover staging (CDNA5).  One descriptor moves a 128x64 bf16 tile
// (row stride 512 elems in global) into LDS with a 16B pad every 128B row,
// matching the LDA=72 layout the WMMA fragment loads expect.
// ---------------------------------------------------------------------------
#if __has_builtin(__builtin_amdgcn_tensor_load_to_lds)
#define USE_TDM 1
#if defined(__clang_major__) && (__clang_major__ >= 23)
#define TDM_CALL(g0, g1) do {                                          \
    i32x4 _z4 = {0, 0, 0, 0};                                          \
    i32x8 _z8 = {0, 0, 0, 0, 0, 0, 0, 0};                              \
    __builtin_amdgcn_tensor_load_to_lds((g0), (g1), _z4, _z4, _z8, 0); \
  } while (0)
#else
#define TDM_CALL(g0, g1) do {                                          \
    i32x4 _z4 = {0, 0, 0, 0};                                          \
    __builtin_amdgcn_tensor_load_to_lds((g0), (g1), _z4, _z4, 0);      \
  } while (0)
#endif

__device__ __forceinline__ void tdm_load_tile(unsigned long long gaddr,
                                              unsigned int lds_off) {
  // ---- D# group 0 ----
  u32x4 g0;
  g0[0] = 1u;                                        // count=1, user mode
  g0[1] = lds_off;                                   // lds_addr (bytes)
  g0[2] = (unsigned int)(gaddr & 0xFFFFFFFFu);       // global_addr[31:0]
  g0[3] = (unsigned int)((gaddr >> 32) & 0x1FFFFFFu) // global_addr[56:32]
        | (2u << 30);                                // type = 2 ("image")
  // ---- D# group 1 ----
  // data_size=1 (2B) @16 | pad_enable @20 | pad_interval=4 (32 DW = 128B) @22
  // | pad_amount=3 (4 DW = 16B) @25
  i32x8 g1;
  g1[0] = (int)((1u << 16) | (1u << 20) | (4u << 22) | (3u << 25));
  g1[1] = (int)(((unsigned)KC & 0xFFFFu) << 16);     // tensor_dim0[15:0] @48
  g1[2] = (int)(((unsigned)BM & 0xFFFFu) << 16);     // dim0 hi=0 | tensor_dim1 lo @80
  g1[3] = (int)(((unsigned)KC & 0xFFFFu) << 16);     // dim1 hi=0 | tile_dim0 @112
  g1[4] = (int)((unsigned)BM);                       // tile_dim1=128, tile_dim2=0
  g1[5] = (int)((unsigned)D_DIM);                    // tensor_dim0_stride lo32
  g1[6] = 0;                                         // stride hi | dim1_stride lo
  g1[7] = 0;                                         // dim1_stride hi
  TDM_CALL(g0, g1);
}
#else
#define USE_TDM 0
#endif

// ---------------------------------------------------------------------------
// Kernel 0: zero the scalar output (harness poisons d_out)
// ---------------------------------------------------------------------------
__global__ void zero_kernel(float* out) {
  if (threadIdx.x == 0 && blockIdx.x == 0) out[0] = 0.0f;
}

// ---------------------------------------------------------------------------
// Kernel 1: row-wise L2 normalize, convert to bf16, emit per-row ||e||^2.
// ---------------------------------------------------------------------------
__global__ __launch_bounds__(256)
void normalize_kernel(const float* __restrict__ emb,
                      unsigned short* __restrict__ ebf,
                      float* __restrict__ sq) {
  __shared__ float sRed[8];
  const int row = blockIdx.x;
  const int tid = threadIdx.x;
  const float* src = emb + (size_t)row * D_DIM;
  float v0 = src[tid];
  float v1 = src[tid + 256];
  float local = v0 * v0 + v1 * v1;
  #pragma unroll
  for (int off = 16; off > 0; off >>= 1) local += __shfl_xor(local, off);
  if ((tid & 31) == 0) sRed[tid >> 5] = local;
  __syncthreads();
  float sumsq = 0.0f;
  #pragma unroll
  for (int i = 0; i < 8; ++i) sumsq += sRed[i];
  const float rn = 1.0f / fmaxf(sqrtf(sumsq), 1e-12f);
  __hip_bfloat16 b0 = __float2bfloat16(v0 * rn);
  __hip_bfloat16 b1 = __float2bfloat16(v1 * rn);
  ebf[(size_t)row * D_DIM + tid]       = *(const unsigned short*)&b0;
  ebf[(size_t)row * D_DIM + tid + 256] = *(const unsigned short*)&b1;
  if (tid == 0) sq[row] = sumsq * rn * rn;   // ~1.0, matches reference sq
}

// ---------------------------------------------------------------------------
// Kernel 2: numerically stable BCE-with-logits, mean over B, atomic into out.
// ---------------------------------------------------------------------------
__global__ __launch_bounds__(256)
void bce_kernel(const float* __restrict__ outputs,
                const float* __restrict__ classes,
                float* __restrict__ out) {
  __shared__ float sRed[8];
  float local = 0.0f;
  for (int i = blockIdx.x * 256 + threadIdx.x; i < B_ROWS; i += gridDim.x * 256) {
    const float o = outputs[i], c = classes[i];
    local += fmaxf(o, 0.0f) - o * c + log1pf(expf(-fabsf(o)));
  }
  #pragma unroll
  for (int off = 16; off > 0; off >>= 1) local += __shfl_xor(local, off);
  if ((threadIdx.x & 31) == 0) sRed[threadIdx.x >> 5] = local;
  __syncthreads();
  if (threadIdx.x == 0) {
    float s = 0.0f;
    #pragma unroll
    for (int i = 0; i < 8; ++i) s += sRed[i];
    atomicAdd(out, s * (1.0f / (float)B_ROWS));
  }
}

// ---------------------------------------------------------------------------
// Kernel 3: fused Gram-matrix + contrastive pair loss over upper-tri blocks.
// 256 threads = 8 waves; block tile 128x128; each wave owns 2x4 WMMA tiles.
// TDM double-buffers the LDS K-chunks; wave 0 drives the DMA descriptors.
// ---------------------------------------------------------------------------
__global__ __launch_bounds__(256, 1)
void gram_loss_kernel(const unsigned short* __restrict__ ebf,
                      const float* __restrict__ sq,
                      const int* __restrict__ ids,
                      const float* __restrict__ cls,
                      float* __restrict__ out) {
  __shared__ __align__(16) unsigned short As[2][BM * LDA];
  __shared__ __align__(16) unsigned short Bs[2][BM * LDA];
  __shared__ float sSqA[BM], sClA[BM], sSqB[BM], sClB[BM];
  __shared__ int   sIdA[BM], sIdB[BM];
  __shared__ float sRed[8];

  const int NB = B_ROWS / BM;                 // 64
  // Linear triangular block index -> (bi, bj), bi <= bj
  int t = blockIdx.x;
  int bi = 0;
  while (t >= NB - bi) { t -= NB - bi; ++bi; }
  const int bj = bi + t;
  const float w = (bi == bj) ? 1.0f : 2.0f;   // symmetry weight
  const int i0 = bi * BM, j0 = bj * BM;

  const int tid = threadIdx.x;
  if (tid < BM) {
    const int i = i0 + tid;
    sSqA[tid] = sq[i]; sIdA[tid] = ids[i]; sClA[tid] = cls[i];
  } else {
    const int j = j0 + (tid - BM);
    sSqB[tid - BM] = sq[j]; sIdB[tid - BM] = ids[j]; sClB[tid - BM] = cls[j];
  }

  const int lane = tid & 31;
  const int wave = tid >> 5;
  const int wm = wave & 3;        // 4 row groups of 32
  const int wn = wave >> 2;       // 2 col groups of 64
  const int h  = lane >> 4;       // lane half (K window select)
  const int ln = lane & 15;       // row/col within 16

  v8f acc[2][4] = {};             // fp32 accumulators, 64 VGPRs

#if USE_TDM
  const unsigned long long gbase = (unsigned long long)(uintptr_t)ebf;
  const unsigned int ldsA[2] = {
      (unsigned int)(uintptr_t)(void*)&As[0][0],
      (unsigned int)(uintptr_t)(void*)&As[1][0] };
  const unsigned int ldsB[2] = {
      (unsigned int)(uintptr_t)(void*)&Bs[0][0],
      (unsigned int)(uintptr_t)(void*)&Bs[1][0] };
  if (wave == 0) {   // uniform wave-level branch: TDM issues once per wave
    tdm_load_tile(gbase + ((unsigned long long)i0 * D_DIM) * 2u, ldsA[0]);
    tdm_load_tile(gbase + ((unsigned long long)j0 * D_DIM) * 2u, ldsB[0]);
  }
#endif

  for (int c = 0; c < NCHUNK; ++c) {
    const int kk = c * KC;
    const int buf = c & 1;
    __syncthreads();              // prev compute done before LDS buffer reuse
#if USE_TDM
    if (wave == 0) {
      if (c + 1 < NCHUNK) {       // prefetch next chunk into the other buffer
        const unsigned long long koff = (unsigned long long)(kk + KC) * 2u;
        tdm_load_tile(gbase + ((unsigned long long)i0 * D_DIM) * 2u + koff,
                      ldsA[buf ^ 1]);
        tdm_load_tile(gbase + ((unsigned long long)j0 * D_DIM) * 2u + koff,
                      ldsB[buf ^ 1]);
        __builtin_amdgcn_s_wait_tensorcnt((short)2);  // current pair landed
      } else {
        __builtin_amdgcn_s_wait_tensorcnt((short)0);
      }
    }
#else
    #pragma unroll
    for (int s = 0; s < SEGS_PER_THREAD; ++s) {
      const int seg = tid + s * 256;
      const int r = seg >> 3;          // row in tile
      const int cseg = seg & 7;        // 16B segment within row
      *(uint4*)&As[buf][r * LDA + cseg * 8] =
          *(const uint4*)(ebf + (size_t)(i0 + r) * D_DIM + kk + cseg * 8);
      *(uint4*)&Bs[buf][r * LDA + cseg * 8] =
          *(const uint4*)(ebf + (size_t)(j0 + r) * D_DIM + kk + cseg * 8);
    }
#endif
    __syncthreads();              // chunk `c` visible to all waves

    const unsigned short* Ab = &As[buf][0];
    const unsigned short* Bb = &Bs[buf][0];
    #pragma unroll
    for (int ks = 0; ks < KC; ks += 32) {
      v16bf afr[2], bfr[4];
      // A (16x32, 16-bit): lane half h=0 -> K {0..7,16..23}; h=1 -> K {8..15,24..31}
      #pragma unroll
      for (int ti = 0; ti < 2; ++ti) {
        const unsigned short* base = Ab + (wm * 32 + ti * 16 + ln) * LDA + ks;
        afr[ti] = load_frag(base + 8 * h, base + 16 + 8 * h);
      }
      // B (32x16, 16-bit): lane half h=0 -> K {0..7,16..23}; h=1 -> K {8..15,24..31}
      #pragma unroll
      for (int tn = 0; tn < 4; ++tn) {
        const unsigned short* base = Bb + (wn * 64 + tn * 16 + ln) * LDA + ks;
        bfr[tn] = load_frag(base + 8 * h, base + 16 + 8 * h);
      }
      #pragma unroll
      for (int ti = 0; ti < 2; ++ti)
        #pragma unroll
        for (int tn = 0; tn < 4; ++tn)
          acc[ti][tn] = __builtin_amdgcn_wmma_f32_16x16x32_bf16(
              false, afr[ti], false, bfr[tn], (short)0, acc[ti][tn],
              false, false);
    }
  }

  // ---- fused epilogue: masked contrastive terms on the Gram tile ----
  float local = 0.0f;
  #pragma unroll
  for (int ti = 0; ti < 2; ++ti) {
    #pragma unroll
    for (int tn = 0; tn < 4; ++tn) {
      const int jl = wn * 64 + tn * 16 + ln;      // column (N = lane % 16)
      const float sqj = sSqB[jl];
      const int   idj = sIdB[jl];
      const float clj = sClB[jl];
      #pragma unroll
      for (int r = 0; r < 8; ++r) {
        const int il = wm * 32 + ti * 16 + h * 8 + r;   // M = r + 8*(lane/16)
        const float g  = acc[ti][tn][r];
        const float d2 = fmaxf(sSqA[il] + sqj - 2.0f * g, 0.0f);
        const bool sid = (sIdA[il] == idj);
        const bool scl = (sClA[il] == clj);
        const float term = (sid && !scl) ? 0.5f * d2
                         : ((!sid && scl) ? fmaxf(0.0f, 0.5f - d2) : 0.0f);
        local += term;
      }
    }
  }
  #pragma unroll
  for (int off = 16; off > 0; off >>= 1) local += __shfl_xor(local, off);
  if (lane == 0) sRed[wave] = local;
  __syncthreads();
  if (tid == 0) {
    float s = 0.0f;
    #pragma unroll
    for (int i = 0; i < 8; ++i) s += sRed[i];
    // loss += ALPHA * 0.5 * weight * sum  ==  0.25 * w * sum
    atomicAdd(out, 0.25f * w * s);
  }
}

// ---------------------------------------------------------------------------
// Launch
// ---------------------------------------------------------------------------
extern "C" void kernel_launch(void* const* d_in, const int* in_sizes, int n_in,
                              void* d_out, int out_size, void* d_ws, size_t ws_size,
                              hipStream_t stream) {
  const float* outputs = (const float*)d_in[0];
  const float* classes = (const float*)d_in[1];
  const float* emb     = (const float*)d_in[2];
  const int*   ids     = (const int*)d_in[3];
  float* out = (float*)d_out;

  unsigned short* ebf = (unsigned short*)d_ws;                  // bf16 e [B, D]
  float* sq = (float*)((char*)d_ws +
                       (size_t)B_ROWS * D_DIM * sizeof(unsigned short));

  zero_kernel<<<1, 64, 0, stream>>>(out);
  normalize_kernel<<<B_ROWS, 256, 0, stream>>>(emb, ebf, sq);
  bce_kernel<<<8, 256, 0, stream>>>(outputs, classes, out);
  const int NB = B_ROWS / BM;
  gram_loss_kernel<<<NB * (NB + 1) / 2, 256, 0, stream>>>(ebf, sq, ids, classes, out);
}
